// NeRF_TCNN_32908039422132
// MI455X (gfx1250) — matrix operand order repaired and checked by reading, not compiled, not run on hardware
//
#include <hip/hip_runtime.h>

// ---------------------------------------------------------------------------
// Fused Instant-NGP NeRF forward for MI455X (gfx1250, wave32, WMMA f16).
// One wave = 16 points. All MLP layers run as v_wmma_f32_16x16x32_f16.
// ---------------------------------------------------------------------------

#define NLEV 16
#define LOG2_T 19
#define T_SIZE (1u << LOG2_T)
#define BOUND_F 10.0f

typedef __attribute__((ext_vector_type(16))) _Float16 v16h;
typedef __attribute__((ext_vector_type(8)))  _Float16 v8h;
typedef __attribute__((ext_vector_type(8)))  float    v8f;

// floor(16 * scale^l), scale = exp2(log2(2048*10/16)/15)
__device__ constexpr int RES_TAB[NLEV] = {
    16, 25, 41, 66, 107, 173, 279, 451,
    726, 1171, 1886, 3040, 4898, 7892, 12716, 20480};

__device__ __forceinline__ uint32_t umin32(uint32_t a, uint32_t b) { return a < b ? a : b; }

#define WMMA_F16(A, B, C) \
    __builtin_amdgcn_wmma_f32_16x16x32_f16(false, (A), false, (B), (short)0, (C), false, false)

// --- one hash-grid level: trilinear interp of 2 features (compile-time res) ---
template <int LVL>
__device__ __forceinline__ float2 enc_level(float x, float y, float z,
                                            const float* __restrict__ tables) {
  constexpr int res = RES_TAB[LVL];
  constexpr bool dense =
      (long long)(res + 1) * (res + 1) * (res + 1) <= (long long)T_SIZE;
  const float fr = (float)res;
  float px = x * fr, py = y * fr, pz = z * fr;
  float fx = floorf(px), fy = floorf(py), fz = floorf(pz);
  float wx = px - fx, wy = py - fy, wz = pz - fz;
  uint32_t ix = (uint32_t)fx, iy = (uint32_t)fy, iz = (uint32_t)fz;
  const float2* tl = reinterpret_cast<const float2*>(tables) + (size_t)LVL * T_SIZE;
  float f0 = 0.f, f1 = 0.f;
#pragma unroll
  for (int c = 0; c < 8; ++c) {
    const uint32_t ox = (c >> 2) & 1, oy = (c >> 1) & 1, oz = c & 1;
    uint32_t cx = umin32(ix + ox, (uint32_t)res);
    uint32_t cy = umin32(iy + oy, (uint32_t)res);
    uint32_t cz = umin32(iz + oz, (uint32_t)res);
    uint32_t idx;
    if constexpr (dense) {
      constexpr uint32_t s = (uint32_t)(res + 1);
      idx = cx + s * cy + s * s * cz;
    } else {
      idx = (cx * 1u) ^ (cy * 2654435761u) ^ (cz * 805459861u);
      idx &= (T_SIZE - 1u);
    }
    float2 g = tl[idx];  // global_load_b64, L2-resident table (64 MB < 192 MB L2)
    float wgt = (ox ? wx : 1.f - wx) * (oy ? wy : 1.f - wy) * (oz ? wz : 1.f - wz);
    f0 = fmaf(g.x, wgt, f0);
    f1 = fmaf(g.y, wgt, f1);
  }
  return make_float2(f0, f1);
}

// A operand (16x32 f16) from per-wave LDS activation tile [16][64], row-major.
// Lane layout: m = lane&15, hs = lane>>4; K in {16hs..16hs+15} (matches B)
__device__ __forceinline__ v16h load_A(const _Float16* actw, int m, int hs, int kbase) {
  const _Float16* p = actw + m * 64 + kbase + hs * 16;
  v8h lo = *(const v8h*)(p);
  v8h hi = *(const v8h*)(p + 8);
  v16h a;
#pragma unroll
  for (int i = 0; i < 8; ++i) { a[i] = lo[i]; a[i + 8] = hi[i]; }
  return a;
}

__global__ __launch_bounds__(256) void nerf_tcnn_fused(
    const float* __restrict__ inp, const float* __restrict__ tables,
    const float* __restrict__ w_s1, const float* __restrict__ w_s2,
    const float* __restrict__ w_c1, const float* __restrict__ w_c2,
    const float* __restrict__ w_c3, float* __restrict__ out, int n) {
  // Weights transposed to [n][k] f16 so B operands are contiguous per lane.
  __shared__ __attribute__((aligned(64))) _Float16 sWs1[64 * 32];  // 32->64
  __shared__ __attribute__((aligned(64))) _Float16 sWs2[32 * 64];  // 64->17 (pad 32)
  __shared__ __attribute__((aligned(64))) _Float16 sWc1[64 * 32];  // 31(pad32)->64
  __shared__ __attribute__((aligned(64))) _Float16 sWc2[64 * 64];  // 64->64
  __shared__ __attribute__((aligned(64))) _Float16 sWc3[16 * 64];  // 64->3 (pad 16)
  __shared__ __attribute__((aligned(64))) _Float16 sAct[8][16 * 64];  // per-wave

  const int tid = threadIdx.x;
  for (int i = tid; i < 64 * 32; i += 256) {
    int nn = i >> 5, kk = i & 31;
    sWs1[i] = (_Float16)w_s1[kk * 64 + nn];
  }
  for (int i = tid; i < 32 * 64; i += 256) {
    int nn = i >> 6, kk = i & 63;
    sWs2[i] = (_Float16)((nn < 17) ? w_s2[kk * 17 + nn] : 0.f);
  }
  for (int i = tid; i < 64 * 32; i += 256) {
    int nn = i >> 5, kk = i & 31;
    sWc1[i] = (_Float16)((kk < 31) ? w_c1[kk * 64 + nn] : 0.f);
  }
  for (int i = tid; i < 64 * 64; i += 256) {
    int nn = i >> 6, kk = i & 63;
    sWc2[i] = (_Float16)w_c2[kk * 64 + nn];
  }
  for (int i = tid; i < 16 * 64; i += 256) {
    int nn = i >> 6, kk = i & 63;
    sWc3[i] = (_Float16)((nn < 3) ? w_c3[kk * 3 + nn] : 0.f);
  }
  __syncthreads();

  const int wave = tid >> 5;
  const int lane = tid & 31;
  const int m = lane & 15;   // matrix row / point within tile, B column
  const int hs = lane >> 4;  // half-wave K/row-group select
  _Float16* act = sAct[wave];

  const int base = blockIdx.x * 128 + wave * 16;
  const int pid = (base + m < n) ? (base + m) : (n - 1);

  const float inv2b = 0.5f / BOUND_F;
  const float x01 = (inp[pid * 6 + 0] + BOUND_F) * inv2b;
  const float y01 = (inp[pid * 6 + 1] + BOUND_F) * inv2b;
  const float z01 = (inp[pid * 6 + 2] + BOUND_F) * inv2b;

  // ---- hash-grid encode straight into the WMMA A-operand register layout ----
  // hs=0 lane owns feature K 0..15  -> levels 0..7
  // hs=1 lane owns feature K 16..31 -> levels 8..15
  v16h aH;
#define ENC(LVL, E)                                         \
  {                                                         \
    float2 f = enc_level<LVL>(x01, y01, z01, tables);       \
    aH[E] = (_Float16)f.x;                                  \
    aH[(E) + 1] = (_Float16)f.y;                            \
  }
  if (hs == 0) {
    ENC(0, 0) ENC(1, 2) ENC(2, 4) ENC(3, 6)
    ENC(4, 8) ENC(5, 10) ENC(6, 12) ENC(7, 14)
  } else {
    ENC(8, 0) ENC(9, 2) ENC(10, 4) ENC(11, 6)
    ENC(12, 8) ENC(13, 10) ENC(14, 12) ENC(15, 14)
  }
#undef ENC

  const v8f cz = {};

  // ---- sigma layer 1: [16x32] @ [32x64] + ReLU -> act[16][64] ----
#pragma unroll
  for (int nt = 0; nt < 4; ++nt) {
    v16h b = *(const v16h*)(sWs1 + (nt * 16 + m) * 32 + hs * 16);
    v8f acc = WMMA_F16(aH, b, cz);
#pragma unroll
    for (int r = 0; r < 8; ++r) {
      float v = acc[r];
      v = v > 0.f ? v : 0.f;
      act[(hs * 8 + r) * 64 + nt * 16 + m] = (_Float16)v;
    }
  }

  // ---- sigma layer 2: [16x64] @ [64x17], no activation ----
  {
    v16h a0 = load_A(act, m, hs, 0);
    v16h a1 = load_A(act, m, hs, 32);
#pragma unroll
    for (int nt = 0; nt < 2; ++nt) {
      v16h b0 = *(const v16h*)(sWs2 + (nt * 16 + m) * 64 + hs * 16);
      v16h b1 = *(const v16h*)(sWs2 + (nt * 16 + m) * 64 + hs * 16 + 32);
      v8f acc = WMMA_F16(a0, b0, cz);
      acc = WMMA_F16(a1, b1, acc);
      const int cg = nt * 16 + m;  // global output column this lane holds
#pragma unroll
      for (int r = 0; r < 8; ++r) {
        const int row = hs * 8 + r;
        const int po = base + row;
        float v = acc[r];
        if (cg == 0) {
          if (po < n) out[po * 5 + 3] = v;              // sigma
        } else if (cg == 1) {
          if (po < n) out[po * 5 + 4] = v;              // logit
        } else if (cg <= 16) {
          act[row * 64 + (cg + 14)] = (_Float16)v;      // geo -> color cols 16..30
        }
      }
    }
  }

  // ---- SH degree-4 encode -> color-input cols 0..15 (+ zero pad col 31) ----
  if (lane < 16) {
    float dx = inp[pid * 6 + 3], dy = inp[pid * 6 + 4], dz = inp[pid * 6 + 5];
    float xx = dx * dx, yy = dy * dy, zz = dz * dz;
    float xy = dx * dy, yz = dy * dz, xz = dx * dz;
    float sh[16];
    sh[0] = 0.28209479177387814f;
    sh[1] = -0.48860251190291987f * dy;
    sh[2] = 0.48860251190291987f * dz;
    sh[3] = -0.48860251190291987f * dx;
    sh[4] = 1.0925484305920792f * xy;
    sh[5] = -1.0925484305920792f * yz;
    sh[6] = 0.94617469575756f * zz - 0.31539156525252f;
    sh[7] = -1.0925484305920792f * xz;
    sh[8] = 0.5462742152960396f * (xx - yy);
    sh[9] = 0.5900435899266435f * dy * (-3.0f * xx + yy);
    sh[10] = 2.890611442640554f * xy * dz;
    sh[11] = 0.4570457994644657f * dy * (1.0f - 5.0f * zz);
    sh[12] = 0.3731763325901154f * dz * (5.0f * zz - 3.0f);
    sh[13] = 0.4570457994644657f * dx * (1.0f - 5.0f * zz);
    sh[14] = 1.445305721320277f * dz * (xx - yy);
    sh[15] = 0.5900435899266435f * dx * (-xx + 3.0f * yy);
    v16h shv;
#pragma unroll
    for (int i = 0; i < 16; ++i) shv[i] = (_Float16)sh[i];
    *(v16h*)(act + m * 64) = shv;
    act[m * 64 + 31] = (_Float16)0.f;
  }

  // ---- color layer 1: [16x32] @ [32x64] + ReLU ----
  {
    v16h a = load_A(act, m, hs, 0);  // loaded to regs before stores overwrite
#pragma unroll
    for (int nt = 0; nt < 4; ++nt) {
      v16h b = *(const v16h*)(sWc1 + (nt * 16 + m) * 32 + hs * 16);
      v8f acc = WMMA_F16(a, b, cz);
#pragma unroll
      for (int r = 0; r < 8; ++r) {
        float v = acc[r];
        v = v > 0.f ? v : 0.f;
        act[(hs * 8 + r) * 64 + nt * 16 + m] = (_Float16)v;
      }
    }
  }

  // ---- color layer 2: [16x64] @ [64x64] + ReLU ----
  {
    v16h a0 = load_A(act, m, hs, 0);
    v16h a1 = load_A(act, m, hs, 32);
#pragma unroll
    for (int nt = 0; nt < 4; ++nt) {
      v16h b0 = *(const v16h*)(sWc2 + (nt * 16 + m) * 64 + hs * 16);
      v16h b1 = *(const v16h*)(sWc2 + (nt * 16 + m) * 64 + hs * 16 + 32);
      v8f acc = WMMA_F16(a0, b0, cz);
      acc = WMMA_F16(a1, b1, acc);
#pragma unroll
      for (int r = 0; r < 8; ++r) {
        float v = acc[r];
        v = v > 0.f ? v : 0.f;
        act[(hs * 8 + r) * 64 + nt * 16 + m] = (_Float16)v;
      }
    }
  }

  // ---- color layer 3: [16x64] @ [64x3] -> rgb ----
  {
    v16h a0 = load_A(act, m, hs, 0);
    v16h a1 = load_A(act, m, hs, 32);
    v16h b0 = *(const v16h*)(sWc3 + m * 64 + hs * 16);
    v16h b1 = *(const v16h*)(sWc3 + m * 64 + hs * 16 + 32);
    v8f acc = WMMA_F16(a0, b0, cz);
    acc = WMMA_F16(a1, b1, acc);
    if (m < 3) {
#pragma unroll
      for (int r = 0; r < 8; ++r) {
        const int po = base + hs * 8 + r;
        if (po < n) out[po * 5 + m] = acc[r];
      }
    }
  }
}

extern "C" void kernel_launch(void* const* d_in, const int* in_sizes, int n_in,
                              void* d_out, int out_size, void* d_ws, size_t ws_size,
                              hipStream_t stream) {
  (void)n_in; (void)out_size; (void)d_ws; (void)ws_size;
  const float* inp    = (const float*)d_in[0];
  const float* tables = (const float*)d_in[1];
  const float* w_s1   = (const float*)d_in[2];
  const float* w_s2   = (const float*)d_in[3];
  const float* w_c1   = (const float*)d_in[4];
  const float* w_c2   = (const float*)d_in[5];
  const float* w_c3   = (const float*)d_in[6];
  float* out = (float*)d_out;
  const int n = in_sizes[0] / 6;
  const int blocks = (n + 127) / 128;  // 128 points per block (8 waves x 16)
  nerf_tcnn_fused<<<blocks, 256, 0, stream>>>(inp, tables, w_s1, w_s2, w_c1,
                                              w_c2, w_c3, out, n);
}
